// GraphTransformerEncoder_43696997269955
// MI455X (gfx1250) — matrix operand order repaired and checked by reading, not compiled, not run on hardware
//
#include <hip/hip_runtime.h>

#define NN 50000
#define GG 50
#define MM 1000          // nodes per graph
#define DD 128
#define HH 8
#define DHH 16
#define LL 3
#define EE 800000
#define BN_EPS 1e-5f

typedef unsigned short ushort_t;
typedef __attribute__((ext_vector_type(16))) __bf16 bf16x16;
typedef __attribute__((ext_vector_type(8)))  float  f32x8;

union BFrag { bf16x16 v; ushort_t u[16]; };

// ---------- helpers ----------
__device__ __forceinline__ ushort_t f2bf(float f) {
  unsigned u = __float_as_uint(f);
  unsigned r = u + 0x7FFFu + ((u >> 16) & 1u);   // round-to-nearest-even
  return (ushort_t)(r >> 16);
}
// order-preserving float<->uint for atomicMax-based segment max
__device__ __forceinline__ unsigned encf(float f) {
  unsigned u = __float_as_uint(f);
  return (u & 0x80000000u) ? ~u : (u | 0x80000000u);
}
__device__ __forceinline__ float decf(unsigned u) {
  unsigned b = (u & 0x80000000u) ? (u ^ 0x80000000u) : ~u;
  return __uint_as_float(b);
}

// ---------- weight convert: W[l][k][n] (f32) -> wt[((l*4+mat)*D + n)*D + k] (bf16, transposed) ----------
__global__ void k_convw(const float* __restrict__ Wq, const float* __restrict__ Wk,
                        const float* __restrict__ Wv, const float* __restrict__ Ws,
                        ushort_t* __restrict__ wt) {
  int i = blockIdx.x * blockDim.x + threadIdx.x;
  if (i >= LL * 4 * DD * DD) return;
  int k   = i & (DD - 1);
  int n   = (i >> 7) & (DD - 1);
  int mat = (i >> 14) & 3;
  int l   = i >> 16;
  const float* W = (mat == 0) ? Wq : (mat == 1) ? Wk : (mat == 2) ? Wv : Ws;
  wt[i] = f2bf(W[((long)l * DD + k) * DD + n]);
}

// ---------- x convert: f32 -> bf16 ----------
__global__ void k_convx(const float* __restrict__ xc, ushort_t* __restrict__ xb) {
  int i = blockIdx.x * blockDim.x + threadIdx.x;
  if (i >= NN * DD) return;
  xb[i] = f2bf(xc[i]);
}

// ---------- fused 4-matrix GEMM: one wave per 16x16 tile, K=128 via 4x wmma_f32_16x16x32_bf16 ----------
__global__ __launch_bounds__(256) void k_gemm(
    const ushort_t* __restrict__ xb, const ushort_t* __restrict__ wt, int layer,
    const float* __restrict__ bq, const float* __restrict__ bk,
    const float* __restrict__ bv, const float* __restrict__ bs,
    float* __restrict__ qo, float* __restrict__ ko,
    float* __restrict__ vo, float* __restrict__ so) {
  const int wave = threadIdx.x >> 5;
  const int lane = threadIdx.x & 31;
  const int half = lane >> 4;
  const int l16  = lane & 15;
  const long tilesM = NN / 16;                       // 3125
  long tile = (long)blockIdx.x * 8 + wave;
  if (tile >= tilesM * 8 * 4) return;                // wave-uniform guard (EXEC all-1s for WMMA)
  int  mat   = (int)(tile / (tilesM * 8));
  long rem   = tile % (tilesM * 8);
  int  tileN = (int)(rem / tilesM);
  long tileM = rem % tilesM;

  // A: row M = lane&15; K = 16h..16h+15 contiguous (h = lane>>4)
  const ushort_t* arow = xb + (tileM * 16 + l16) * DD + 16 * half;
  // B: col N = lane&15; K = 16h..16h+15 contiguous (weights pre-transposed: wt[n][k])
  const ushort_t* brow = wt + (((long)layer * 4 + mat) * DD + (tileN * 16 + l16)) * (long)DD + 16 * half;

  f32x8 c = {};
#pragma unroll
  for (int kk = 0; kk < DD; kk += 32) {
    BFrag a, b;
#pragma unroll
    for (int i = 0; i < 8; ++i) {
      a.u[i]     = arow[kk + i];
      a.u[8 + i] = arow[kk + 8 + i];
    }
#pragma unroll
    for (int i = 0; i < 16; ++i) b.u[i] = brow[kk + i];
    c = __builtin_amdgcn_wmma_f32_16x16x32_bf16(false, a.v, false, b.v, (short)0, c, false, false);
  }

  const float* bias; float* outp;
  if      (mat == 0) { bias = bq; outp = qo; }
  else if (mat == 1) { bias = bk; outp = ko; }
  else if (mat == 2) { bias = bv; outp = vo; }
  else               { bias = bs; outp = so; }
  int col = tileN * 16 + l16;
  float bcol = bias[col];
#pragma unroll
  for (int r = 0; r < 8; ++r) {                      // C/D: M = r + 8h, N = lane&15
    long row = tileM * 16 + r + 8 * half;
    outp[row * DD + col] = c[r] + bcol;
  }
}

// ---------- pass 1: per-(edge,head) logits + segment max via encoded atomicMax ----------
__global__ __launch_bounds__(256) void k_logits(const int* __restrict__ ei,
    const float* __restrict__ qb, const float* __restrict__ kb,
    float* __restrict__ logits, unsigned* __restrict__ mx) {
  int tid = blockIdx.x * blockDim.x + threadIdx.x;
  if (tid >= EE * HH) return;
  int e = tid >> 3, h = tid & 7;
  int s = ei[e], d = ei[EE + e];
  const float4* qp = (const float4*)(qb + (long)d * DD + h * DHH);
  const float4* kp = (const float4*)(kb + (long)s * DD + h * DHH);
  float acc = 0.f;
#pragma unroll
  for (int j = 0; j < 4; ++j) {
    float4 a = qp[j], b = kp[j];
    acc += a.x * b.x + a.y * b.y + a.z * b.z + a.w * b.w;
  }
  float lg = acc * 0.25f;                            // 1/sqrt(DH)
  logits[tid] = lg;
  atomicMax(&mx[(long)d * HH + h], encf(lg));
}

// ---------- pass 2: ex = exp(l - max); segment sum ----------
__global__ __launch_bounds__(256) void k_exp(const int* __restrict__ ei,
    float* __restrict__ logits, const unsigned* __restrict__ mx,
    float* __restrict__ denom) {
  int tid = blockIdx.x * blockDim.x + threadIdx.x;
  if (tid >= EE * HH) return;
  int e = tid >> 3, h = tid & 7;
  int d = ei[EE + e];
  float m  = decf(mx[(long)d * HH + h]);
  float ex = __expf(logits[tid] - m);
  logits[tid] = ex;
  atomicAdd(&denom[(long)d * HH + h], ex);
}

// ---------- pass 3: acc[dst] += v[src] * alpha ----------
__global__ __launch_bounds__(256) void k_scatter(const int* __restrict__ ei,
    const float* __restrict__ exb, const float* __restrict__ denom,
    const float* __restrict__ vb, float* __restrict__ acc) {
  int tid = blockIdx.x * blockDim.x + threadIdx.x;
  if (tid >= EE * 32) return;
  int e = tid >> 5, lane = tid & 31;
  int c0 = lane * 4, h = c0 >> 4;
  int s = ei[e], d = ei[EE + e];
  float alpha = exb[e * HH + h] / denom[(long)d * HH + h];
  float4 vv = *(const float4*)(vb + (long)s * DD + c0);
  float* ap = acc + (long)d * DD + c0;
  atomicAdd(ap + 0, vv.x * alpha);
  atomicAdd(ap + 1, vv.y * alpha);
  atomicAdd(ap + 2, vv.z * alpha);
  atomicAdd(ap + 3, vv.w * alpha);
}

// ---------- combine: pre = x + agg + skip (in place in acc) + column sums for BN ----------
__global__ __launch_bounds__(256) void k_combine(const float* __restrict__ xc,
    const float* __restrict__ sb, float* __restrict__ acc,
    float* __restrict__ colsum, float* __restrict__ colsumsq) {
  __shared__ float sm1[256], sm2[256];
  float s1 = 0.f, s2 = 0.f;
  int stride = gridDim.x * blockDim.x;               // multiple of 128 -> column fixed per thread
  for (int i = blockIdx.x * blockDim.x + threadIdx.x; i < NN * DD; i += stride) {
    float p = xc[i] + acc[i] + sb[i];
    acc[i] = p;
    s1 += p; s2 += p * p;
  }
  sm1[threadIdx.x] = s1; sm2[threadIdx.x] = s2;
  __syncthreads();
  if (threadIdx.x < 128) {                           // col == threadIdx.x (blockDim=256, 256%128==0)
    atomicAdd(&colsum[threadIdx.x],   sm1[threadIdx.x] + sm1[threadIdx.x + 128]);
    atomicAdd(&colsumsq[threadIdx.x], sm2[threadIdx.x] + sm2[threadIdx.x + 128]);
  }
}

// ---------- BN apply ----------
__global__ __launch_bounds__(256) void k_bn(const float* __restrict__ pre,
    const float* __restrict__ colsum, const float* __restrict__ colsumsq,
    const float* __restrict__ gamma, const float* __restrict__ beta,
    float* __restrict__ xc) {
  int i = blockIdx.x * blockDim.x + threadIdx.x;
  if (i >= NN * DD) return;
  int col = i & (DD - 1);
  float mean = colsum[col] * (1.0f / NN);
  float var  = colsumsq[col] * (1.0f / NN) - mean * mean;
  xc[i] = (pre[i] - mean) * rsqrtf(var + BN_EPS) * gamma[col] + beta[col];
}

// ---------- graph mean pool (sorted, equal-sized graphs) ----------
__global__ void k_pool(const float* __restrict__ xc, float* __restrict__ out) {
  int g = blockIdx.x, col = threadIdx.x;
  float s = 0.f;
  for (int r = 0; r < MM; ++r) s += xc[((long)g * MM + r) * DD + col];
  out[g * DD + col] = s * (1.0f / MM);
}

extern "C" void kernel_launch(void* const* d_in, const int* in_sizes, int n_in,
                              void* d_out, int out_size, void* d_ws, size_t ws_size,
                              hipStream_t stream) {
  const float* x   = (const float*)d_in[0];
  const int*   ei  = (const int*)d_in[1];     // edge_index [2,E]
  // d_in[2] = batch (unused: sorted, equal-sized graphs)
  const float* Wq  = (const float*)d_in[3];
  const float* bq  = (const float*)d_in[4];
  const float* Wk  = (const float*)d_in[5];
  const float* bk  = (const float*)d_in[6];
  const float* Wv  = (const float*)d_in[7];
  const float* bv  = (const float*)d_in[8];
  const float* Ws  = (const float*)d_in[9];
  const float* bs  = (const float*)d_in[10];
  const float* gamma = (const float*)d_in[11];
  const float* beta  = (const float*)d_in[12];
  float* out = (float*)d_out;

  // carve workspace
  char* w = (char*)d_ws;
  size_t off = 0;
  auto carve = [&](size_t bytes) -> void* {
    void* p = w + off;
    off = (off + bytes + 255) & ~(size_t)255;
    return p;
  };
  float*    qb     = (float*)carve(sizeof(float) * NN * DD);   // reused as agg accumulator
  float*    kb     = (float*)carve(sizeof(float) * NN * DD);
  float*    vb     = (float*)carve(sizeof(float) * NN * DD);
  float*    sb     = (float*)carve(sizeof(float) * NN * DD);
  float*    exb    = (float*)carve(sizeof(float) * EE * HH);   // logits, then ex
  unsigned* mx     = (unsigned*)carve(sizeof(unsigned) * NN * HH);
  float*    denom  = (float*)carve(sizeof(float) * NN * HH);
  float*    xc     = (float*)carve(sizeof(float) * NN * DD);
  ushort_t* xb     = (ushort_t*)carve(sizeof(ushort_t) * NN * DD);
  ushort_t* wt     = (ushort_t*)carve(sizeof(ushort_t) * LL * 4 * DD * DD);
  float*    colsum = (float*)carve(sizeof(float) * 256);       // colsum[128] | colsumsq[128]
  float*    colsumsq = colsum + 128;

  const int B = 256;
  hipMemcpyAsync(xc, x, sizeof(float) * NN * DD, hipMemcpyDeviceToDevice, stream);
  k_convw<<<(LL * 4 * DD * DD + B - 1) / B, B, 0, stream>>>(Wq, Wk, Wv, Ws, wt);

  for (int l = 0; l < LL; ++l) {
    k_convx<<<(NN * DD + B - 1) / B, B, 0, stream>>>(xc, xb);
    hipMemsetAsync(mx,    0, sizeof(unsigned) * NN * HH, stream);
    hipMemsetAsync(denom, 0, sizeof(float) * NN * HH, stream);
    hipMemsetAsync(colsum, 0, sizeof(float) * 256, stream);

    k_gemm<<<(NN / 16) * 8 * 4 / 8, B, 0, stream>>>(
        xb, wt, l, bq + l * DD, bk + l * DD, bv + l * DD, bs + l * DD, qb, kb, vb, sb);

    k_logits<<<(EE * HH + B - 1) / B, B, 0, stream>>>(ei, qb, kb, exb, mx);
    hipMemsetAsync(qb, 0, sizeof(float) * NN * DD, stream);   // qb becomes agg accumulator
    k_exp<<<(EE * HH + B - 1) / B, B, 0, stream>>>(ei, exb, mx, denom);
    k_scatter<<<(EE * 32 + B - 1) / B, B, 0, stream>>>(ei, exb, denom, vb, qb);

    k_combine<<<256, B, 0, stream>>>(xc, sb, qb, colsum, colsumsq);
    k_bn<<<(NN * DD + B - 1) / B, B, 0, stream>>>(qb, colsum, colsumsq,
                                                  gamma + l * DD, beta + l * DD, xc);
  }

  hipMemcpyAsync(out, xc, sizeof(float) * NN * DD, hipMemcpyDeviceToDevice, stream);
  k_pool<<<GG, DD, 0, stream>>>(xc, out + (size_t)NN * DD);
}
